// GAT_12412455486108
// MI455X (gfx1250) — matrix operand order repaired and checked by reading, not compiled, not run on hardware
//
#include <hip/hip_runtime.h>
#include <hip/hip_bf16.h>
#include <math.h>

#define N_NODES   100000
#define N_EDGES   1600000
#define IN_DIM    128
#define HID       32
#define HEADS     4
#define D0        128      // HEADS*HID
#define OUT_DIM   40
#define NEG_SLOPE 0.2f
#define BN_EPS    1e-5f

typedef __attribute__((ext_vector_type(16))) __bf16 v16bf;
typedef __attribute__((ext_vector_type(8)))  float  v8f;

__device__ __forceinline__ float lrelu(float x) { return x > 0.f ? x : NEG_SLOPE * x; }

// monotone float <-> uint encoding so unsigned atomicMax == float max
__device__ __forceinline__ unsigned fenc(float f) {
    unsigned u = __float_as_uint(f);
    return (u & 0x80000000u) ? ~u : (u | 0x80000000u);
}
__device__ __forceinline__ float fdec(unsigned u) {
    return __uint_as_float((u & 0x80000000u) ? (u ^ 0x80000000u) : ~u);
}

// ---------------------------------------------------------------------------
// 0. zero small accumulator buffer
__global__ void k_zero(float* p, int n) {
    int i = blockIdx.x * blockDim.x + threadIdx.x;
    if (i < n) p[i] = 0.f;
}

// ---------------------------------------------------------------------------
// 1. GEMM0: H = X(N x 128) @ W0(128 x 128) + b0, bf16 WMMA, f32 accum.
//    Block: 256 threads (8 waves) -> 32 rows x 128 cols.
//    W0 staged bf16 in LDS (transposed, padded); X tile staged bf16 in LDS.
//    Wave w owns 16-col tile w and computes 2 M-tiles vs one shared B frag.
#define W0_LDP 136   // padded K stride for W in LDS (bank spread)
#define X_LDP  132   // padded K stride for X in LDS
__global__ void k_gemm0(const float* __restrict__ X, const float* __restrict__ W,
                        const float* __restrict__ bias, float* __restrict__ H) {
    __shared__ __bf16 Wl[D0 * W0_LDP];   // [n][k]
    __shared__ __bf16 Xl[32 * X_LDP];    // [m][k]
    const int tid = threadIdx.x;
    const int m0  = blockIdx.x * 32;

    // stage W0 -> LDS (bf16, transposed): 16384 elems, 64 per thread, coalesced reads
#pragma unroll 4
    for (int j = 0; j < 64; ++j) {
        int i = tid + 256 * j;
        int k = i >> 7, n = i & 127;
        Wl[n * W0_LDP + k] = (__bf16)W[i];
    }
    // stage X tile -> LDS: 4096 elems, 16 per thread, coalesced reads
#pragma unroll 4
    for (int j = 0; j < 16; ++j) {
        int i = tid + 256 * j;
        int r = i >> 7, c = i & 127;
        Xl[r * X_LDP + c] = (__bf16)X[(size_t)(m0 + r) * IN_DIM + c];
    }
    __syncthreads();

    const int wave = tid >> 5;
    const int lane = tid & 31;
    const int n0 = wave * 16;
    const int g  = lane >> 4;     // lane half (K group select)
    const int lm = lane & 15;

    v8f acc0 = {}, acc1 = {};
#pragma unroll
    for (int kc = 0; kc < IN_DIM; kc += 32) {
        v16bf a0, a1, b;
        int kb = kc + (g << 3);
#pragma unroll
        for (int e = 0; e < 16; ++e)           // B 32x16 from LDS (ISA per-lane K layout)
            b[e] = Wl[(n0 + lm) * W0_LDP + kb + e + ((e >> 3) << 3)];
#pragma unroll
        for (int r = 0; r < 8; ++r) {          // A 16x32: ISA per-lane K layout
            int k = kc + ((r >> 2) << 4) + (g << 3) + ((r & 3) << 1);
            a0[2 * r]     = Xl[lm * X_LDP + k];
            a0[2 * r + 1] = Xl[lm * X_LDP + k + 1];
            a1[2 * r]     = Xl[(16 + lm) * X_LDP + k];
            a1[2 * r + 1] = Xl[(16 + lm) * X_LDP + k + 1];
        }
        acc0 = __builtin_amdgcn_wmma_f32_16x16x32_bf16(false, a0, false, b,
                                                       (short)0, acc0, false, false);
        acc1 = __builtin_amdgcn_wmma_f32_16x16x32_bf16(false, a1, false, b,
                                                       (short)0, acc1, false, false);
    }
    int col = n0 + lm;
    float bv = bias[col];
#pragma unroll
    for (int r = 0; r < 8; ++r) {              // C/D: row = r + 8*g
        int row = m0 + (g << 3) + r;
        H[(size_t)row * D0 + col]        = acc0[r] + bv;
        H[(size_t)(row + 16) * D0 + col] = acc1[r] + bv;
    }
}

// ---------------------------------------------------------------------------
// 2. per-(node,head) attention logits for layer 0
__global__ void k_alpha0(const float* __restrict__ H, const float* __restrict__ a_src,
                         const float* __restrict__ a_dst, float* __restrict__ as,
                         float* __restrict__ ad) {
    int i = blockIdx.x * blockDim.x + threadIdx.x;
    if (i >= N_NODES * HEADS) return;
    int node = i >> 2, h = i & 3;
    const float* hp = H + (size_t)node * D0 + h * HID;
    float s = 0.f, d = 0.f;
#pragma unroll 8
    for (int c = 0; c < HID; ++c) {
        float v = hp[c];
        s += v * a_src[h * HID + c];
        d += v * a_dst[h * HID + c];
    }
    as[i] = s; ad[i] = d;
}

// 3. init segment max with self-loop logit
__global__ void k_initmax0(const float* __restrict__ as, const float* __restrict__ ad,
                           unsigned* __restrict__ mu) {
    int i = blockIdx.x * blockDim.x + threadIdx.x;
    if (i >= N_NODES * HEADS) return;
    mu[i] = fenc(lrelu(as[i] + ad[i]));
}

// 4. edge atomic max, thread per (edge, head)
__global__ void k_edgemax0(const int* __restrict__ ei, const float* __restrict__ as,
                           const float* __restrict__ ad, unsigned* __restrict__ mu) {
    int t = blockIdx.x * blockDim.x + threadIdx.x;
    if (t >= N_EDGES * HEADS) return;
    int e = t >> 2, h = t & 3;
    int s = ei[e], d = ei[N_EDGES + e];
    float v = lrelu(as[s * HEADS + h] + ad[d * HEADS + h]);
    atomicMax(&mu[d * HEADS + h], fenc(v));
}

// 5. self-loop contribution: initialize den and agg
__global__ void k_selfinit0(const float* __restrict__ H, const float* __restrict__ as,
                            const float* __restrict__ ad, const unsigned* __restrict__ mu,
                            float* __restrict__ den, float* __restrict__ agg) {
    int i = blockIdx.x * blockDim.x + threadIdx.x;     // node*128 + c
    if (i >= N_NODES * D0) return;
    int node = i >> 7, c = i & 127, h = c >> 5;
    int nh = node * HEADS + h;
    float w = __expf(lrelu(as[nh] + ad[nh]) - fdec(mu[nh]));
    if ((c & 31) == 0) den[nh] = w;
    agg[i] = w * H[i];
}

// 6. wave-per-edge weighted scatter-aggregate (grid-stride, prefetch next row).
//    Lane handles 4 channels (one head).
__global__ void k_edgeagg0(const int* __restrict__ ei, const float* __restrict__ as,
                           const float* __restrict__ ad, const unsigned* __restrict__ mu,
                           const float* __restrict__ H, float* __restrict__ den,
                           float* __restrict__ agg) {
    int wid  = (blockIdx.x * blockDim.x + threadIdx.x) >> 5;
    int lane = threadIdx.x & 31;
    int nw   = (gridDim.x * blockDim.x) >> 5;
    int h    = lane >> 3;                               // 8 lanes per head
    for (int e = wid; e < N_EDGES; e += nw) {
        if (e + nw < N_EDGES) {                         // prefetch next src row (L2 warm)
            int sn = ei[e + nw];
            __builtin_prefetch(H + (size_t)sn * D0 + lane * 4, 0, 1);
        }
        int s = ei[e], d = ei[N_EDGES + e];
        float w = __expf(lrelu(as[s * HEADS + h] + ad[d * HEADS + h]) -
                         fdec(mu[d * HEADS + h]));
        if ((lane & 7) == 0) atomicAdd(&den[d * HEADS + h], w);
        float4 hv = *(const float4*)(H + (size_t)s * D0 + lane * 4);
        float* ap = agg + (size_t)d * D0 + lane * 4;
        atomicAdd(ap + 0, w * hv.x);
        atomicAdd(ap + 1, w * hv.y);
        atomicAdd(ap + 2, w * hv.z);
        atomicAdd(ap + 3, w * hv.w);
    }
}

// 7. normalize + bias, write activated-feature buffer, accumulate BN stats
#define BN_CHUNK 256
__global__ void k_bnstats(const float* __restrict__ agg, const float* __restrict__ den,
                          const float* __restrict__ b0, float* __restrict__ hact,
                          float* __restrict__ sums) {
    int c  = threadIdx.x;                 // 128 threads
    int h  = c >> 5;
    int n0 = blockIdx.x * BN_CHUNK;
    int n1 = min(n0 + BN_CHUNK, N_NODES);
    float bias = b0[c], s = 0.f, ss = 0.f;
    for (int i = n0; i < n1; ++i) {
        float v = agg[(size_t)i * D0 + c] / den[i * HEADS + h] + bias;
        hact[(size_t)i * D0 + c] = v;
        s += v; ss += v * v;
    }
    atomicAdd(&sums[c], s);
    atomicAdd(&sums[D0 + c], ss);
}

// 8. BN finalize: sums -> (scale, shift) in place
__global__ void k_bnfinal(float* __restrict__ sums, const float* __restrict__ gamma,
                          const float* __restrict__ beta) {
    int c = threadIdx.x;
    float mu  = sums[c] / (float)N_NODES;
    float var = sums[D0 + c] / (float)N_NODES - mu * mu;
    float sc  = gamma[c] * rsqrtf(var + BN_EPS);
    sums[c]      = sc;
    sums[D0 + c] = beta[c] - mu * sc;
}

// 9. apply BN + ELU in place
__global__ void k_bnapply(float* __restrict__ hact, const float* __restrict__ sums) {
    int i = blockIdx.x * blockDim.x + threadIdx.x;
    if (i >= N_NODES * D0) return;
    int c = i & 127;
    float v = hact[i] * sums[c] + sums[D0 + c];
    hact[i] = v > 0.f ? v : expm1f(v);
}

// ---------------------------------------------------------------------------
// 10. GEMM1: H1 = Hact(N x 128) @ W1(128 x 40) + b1.
//     Block: 96 threads (3 waves) -> 32 rows x 40 cols (3 masked 16-col tiles).
#define W1_COLS 48   // OUT_DIM padded to 3 WMMA tiles
__global__ void k_gemm1(const float* __restrict__ X, const float* __restrict__ W,
                        const float* __restrict__ bias, float* __restrict__ H1) {
    __shared__ __bf16 Wl[W1_COLS * W0_LDP];  // [n][k]
    __shared__ __bf16 Xl[32 * X_LDP];        // [m][k]
    const int tid = threadIdx.x;
    const int m0  = blockIdx.x * 32;

    // stage W1 -> LDS (bf16, transposed, zero-padded cols): 6144 elems, 64/thread
#pragma unroll 4
    for (int j = 0; j < 64; ++j) {
        int i = tid + 96 * j;                 // 96*64 = 6144
        int k = i / W1_COLS, n = i - k * W1_COLS;
        Wl[n * W0_LDP + k] = (__bf16)(n < OUT_DIM ? W[(size_t)k * OUT_DIM + n] : 0.f);
    }
    // stage X tile -> LDS: 4096 elems
    for (int j = 0; j < 43; ++j) {
        int i = tid + 96 * j;
        if (i < 32 * D0) {
            int r = i >> 7, c = i & 127;
            Xl[r * X_LDP + c] = (__bf16)X[(size_t)(m0 + r) * D0 + c];
        }
    }
    __syncthreads();

    const int wave = tid >> 5;                // 0..2 -> col tile
    const int lane = tid & 31;
    const int n0 = wave * 16;
    const int g  = lane >> 4;
    const int lm = lane & 15;
    const int col = n0 + lm;

    v8f acc0 = {}, acc1 = {};
#pragma unroll
    for (int kc = 0; kc < D0; kc += 32) {
        v16bf a0, a1, b;
        int kb = kc + (g << 3);
#pragma unroll
        for (int e = 0; e < 16; ++e)
            b[e] = Wl[col * W0_LDP + kb + e + ((e >> 3) << 3)];
#pragma unroll
        for (int r = 0; r < 8; ++r) {
            int k = kc + ((r >> 2) << 4) + (g << 3) + ((r & 3) << 1);
            a0[2 * r]     = Xl[lm * X_LDP + k];
            a0[2 * r + 1] = Xl[lm * X_LDP + k + 1];
            a1[2 * r]     = Xl[(16 + lm) * X_LDP + k];
            a1[2 * r + 1] = Xl[(16 + lm) * X_LDP + k + 1];
        }
        acc0 = __builtin_amdgcn_wmma_f32_16x16x32_bf16(false, a0, false, b,
                                                       (short)0, acc0, false, false);
        acc1 = __builtin_amdgcn_wmma_f32_16x16x32_bf16(false, a1, false, b,
                                                       (short)0, acc1, false, false);
    }
    if (col < OUT_DIM) {
        float bv = bias[col];
#pragma unroll
        for (int r = 0; r < 8; ++r) {
            int row = m0 + (g << 3) + r;
            H1[(size_t)row * OUT_DIM + col]        = acc0[r] + bv;
            H1[(size_t)(row + 16) * OUT_DIM + col] = acc1[r] + bv;
        }
    }
}

// 11. layer-1 logits (1 head, 40 ch), thread per node
__global__ void k_alpha1(const float* __restrict__ H1, const float* __restrict__ a_src,
                         const float* __restrict__ a_dst, float* __restrict__ as,
                         float* __restrict__ ad) {
    int n = blockIdx.x * blockDim.x + threadIdx.x;
    if (n >= N_NODES) return;
    const float* hp = H1 + (size_t)n * OUT_DIM;
    float s = 0.f, d = 0.f;
#pragma unroll 8
    for (int c = 0; c < OUT_DIM; ++c) {
        float v = hp[c];
        s += v * a_src[c]; d += v * a_dst[c];
    }
    as[n] = s; ad[n] = d;
}

__global__ void k_initmax1(const float* __restrict__ as, const float* __restrict__ ad,
                           unsigned* __restrict__ mu) {
    int n = blockIdx.x * blockDim.x + threadIdx.x;
    if (n >= N_NODES) return;
    mu[n] = fenc(lrelu(as[n] + ad[n]));
}

__global__ void k_edgemax1(const int* __restrict__ ei, const float* __restrict__ as,
                           const float* __restrict__ ad, unsigned* __restrict__ mu) {
    int e = blockIdx.x * blockDim.x + threadIdx.x;
    if (e >= N_EDGES) return;
    int s = ei[e], d = ei[N_EDGES + e];
    atomicMax(&mu[d], fenc(lrelu(as[s] + ad[d])));
}

// self-loop init of den1 and d_out
__global__ void k_selfinit1(const float* __restrict__ H1, const float* __restrict__ as,
                            const float* __restrict__ ad, const unsigned* __restrict__ mu,
                            float* __restrict__ den, float* __restrict__ out) {
    int i = blockIdx.x * blockDim.x + threadIdx.x;     // node*40 + c
    if (i >= N_NODES * OUT_DIM) return;
    int node = i / OUT_DIM, c = i - node * OUT_DIM;
    float w = __expf(lrelu(as[node] + ad[node]) - fdec(mu[node]));
    if (c == 0) den[node] = w;
    out[i] = w * H1[i];
}

// wave-per-edge scatter for layer 1 (grid-stride, prefetch next row, 40 channels)
__global__ void k_edgeagg1(const int* __restrict__ ei, const float* __restrict__ as,
                           const float* __restrict__ ad, const unsigned* __restrict__ mu,
                           const float* __restrict__ H1, float* __restrict__ den,
                           float* __restrict__ out) {
    int wid  = (blockIdx.x * blockDim.x + threadIdx.x) >> 5;
    int lane = threadIdx.x & 31;
    int nw   = (gridDim.x * blockDim.x) >> 5;
    for (int e = wid; e < N_EDGES; e += nw) {
        if (e + nw < N_EDGES) {
            int sn = ei[e + nw];
            __builtin_prefetch(H1 + (size_t)sn * OUT_DIM + lane, 0, 1);
        }
        int s = ei[e], d = ei[N_EDGES + e];
        float w = __expf(lrelu(as[s] + ad[d]) - fdec(mu[d]));
        if (lane == 0) atomicAdd(&den[d], w);
        const float* hs = H1 + (size_t)s * OUT_DIM;
        float* op = out + (size_t)d * OUT_DIM;
        atomicAdd(&op[lane], w * hs[lane]);
        if (lane < OUT_DIM - 32) atomicAdd(&op[32 + lane], w * hs[32 + lane]);
    }
}

// final normalize + bias (in place on d_out)
__global__ void k_final1(float* __restrict__ out, const float* __restrict__ den,
                         const float* __restrict__ b1) {
    int i = blockIdx.x * blockDim.x + threadIdx.x;
    if (i >= N_NODES * OUT_DIM) return;
    int node = i / OUT_DIM, c = i - node * OUT_DIM;
    out[i] = out[i] / den[node] + b1[c];
}

// ---------------------------------------------------------------------------
extern "C" void kernel_launch(void* const* d_in, const int* in_sizes, int n_in,
                              void* d_out, int out_size, void* d_ws, size_t ws_size,
                              hipStream_t stream) {
    const float* x      = (const float*)d_in[0];
    const int*   ei     = (const int*)d_in[1];
    const float* W0     = (const float*)d_in[2];
    const float* a_src0 = (const float*)d_in[3];
    const float* a_dst0 = (const float*)d_in[4];
    const float* b0     = (const float*)d_in[5];
    const float* gamma0 = (const float*)d_in[6];
    const float* beta0  = (const float*)d_in[7];
    const float* W1     = (const float*)d_in[8];
    const float* a_src1 = (const float*)d_in[9];
    const float* a_dst1 = (const float*)d_in[10];
    const float* b1     = (const float*)d_in[11];
    float* out = (float*)d_out;

    // workspace layout (floats)
    float* ws = (float*)d_ws;
    size_t off = 0;
    float*    h0   = ws + off; off += (size_t)N_NODES * D0;      // lin-out L0, reused as hact
    float*    agg0 = ws + off; off += (size_t)N_NODES * D0;
    float*    as0  = ws + off; off += (size_t)N_NODES * HEADS;
    float*    ad0  = ws + off; off += (size_t)N_NODES * HEADS;
    float*    den0 = ws + off; off += (size_t)N_NODES * HEADS;
    unsigned* m0u  = (unsigned*)(ws + off); off += (size_t)N_NODES * HEADS;
    float*    h1   = ws + off; off += (size_t)N_NODES * OUT_DIM;
    float*    as1  = ws + off; off += (size_t)N_NODES;
    float*    ad1  = ws + off; off += (size_t)N_NODES;
    float*    den1 = ws + off; off += (size_t)N_NODES;
    unsigned* m1u  = (unsigned*)(ws + off); off += (size_t)N_NODES;
    float*    sums = ws + off; off += 2 * D0;

    const int B = 256;
    const int EDGE_BLOCKS = 2048;     // persistent waves for scatter kernels
    // layer 0
    k_zero<<<1, B, 0, stream>>>(sums, 2 * D0);
    k_gemm0<<<N_NODES / 32, 256, 0, stream>>>(x, W0, b0, h0);
    k_alpha0<<<(N_NODES * HEADS + B - 1) / B, B, 0, stream>>>(h0, a_src0, a_dst0, as0, ad0);
    k_initmax0<<<(N_NODES * HEADS + B - 1) / B, B, 0, stream>>>(as0, ad0, m0u);
    k_edgemax0<<<(N_EDGES * HEADS + B - 1) / B, B, 0, stream>>>(ei, as0, ad0, m0u);
    k_selfinit0<<<(N_NODES * D0 + B - 1) / B, B, 0, stream>>>(h0, as0, ad0, m0u, den0, agg0);
    k_edgeagg0<<<EDGE_BLOCKS, B, 0, stream>>>(ei, as0, ad0, m0u, h0, den0, agg0);
    k_bnstats<<<(N_NODES + BN_CHUNK - 1) / BN_CHUNK, D0, 0, stream>>>(agg0, den0, b0, h0, sums);
    k_bnfinal<<<1, D0, 0, stream>>>(sums, gamma0, beta0);
    k_bnapply<<<(N_NODES * D0 + B - 1) / B, B, 0, stream>>>(h0, sums);
    // layer 1
    k_gemm1<<<N_NODES / 32, 96, 0, stream>>>(h0, W1, b1, h1);
    k_alpha1<<<(N_NODES + B - 1) / B, B, 0, stream>>>(h1, a_src1, a_dst1, as1, ad1);
    k_initmax1<<<(N_NODES + B - 1) / B, B, 0, stream>>>(as1, ad1, m1u);
    k_edgemax1<<<(N_EDGES + B - 1) / B, B, 0, stream>>>(ei, as1, ad1, m1u);
    k_selfinit1<<<(N_NODES * OUT_DIM + B - 1) / B, B, 0, stream>>>(h1, as1, ad1, m1u, den1, out);
    k_edgeagg1<<<EDGE_BLOCKS, B, 0, stream>>>(ei, as1, ad1, m1u, h1, den1, out);
    k_final1<<<(N_NODES * OUT_DIM + B - 1) / B, B, 0, stream>>>(out, den1, b1);
}
